// SphericalConv_83313775608514
// MI455X (gfx1250) — matrix operand order repaired and checked by reading, not compiled, not run on hardware
//
#include <hip/hip_runtime.h>

// ---------------- problem constants ----------------
#define L_BW 32
#define N_G  64
#define M_M  63            // 2L-1
#define NB   8
#define CIN  16
#define COUT 16
#define LM   2016          // L_BW * M_M   (= 126*16 = 63*32)
#define JK   4096          // N_G * N_G    (= 256*16 = 128*32)
#define BC   128           // NB * CIN
#define BD   128           // NB * COUT
#define PI_F 3.14159265358979323846f

typedef __attribute__((ext_vector_type(16))) __bf16 v16bf;
typedef __attribute__((ext_vector_type(8)))  __bf16 v8bf;
typedef __attribute__((ext_vector_type(8)))  float  v8f;

union AB32 { v16bf v; v8bf h[2]; };

// ---------------- CDNA5 Tensor Data Mover support ----------------
#ifndef __has_builtin
#define __has_builtin(x) 0
#endif
#if __has_builtin(__builtin_amdgcn_tensor_load_to_lds) && \
    __has_builtin(__builtin_amdgcn_s_wait_tensorcnt)
#define USE_TDM 1
#else
#define USE_TDM 0
#endif

#if USE_TDM
typedef unsigned int u32;
typedef __attribute__((ext_vector_type(4))) u32 u32x4;
typedef __attribute__((ext_vector_type(4))) int i32x4;
typedef __attribute__((ext_vector_type(8))) int i32x8;

// Issue one TDM 2D-tile load: tile_rows x tile_k bf16 elements, row pitch
// row_stride_elems, from global tile start gtile into LDS at lds_off.
// D# packing per CDNA5 ISA ch.8 (group0 128b, group1 256b, groups2/3 zeros-ish).
// This toolchain (clang-23 / therock-10.0) exposes the 6-arg builtin:
//   (uint32x4 g0, int32x8 g1, int32x4 g2, int32x4 g3, int32x8 extra, i32 cpol)
__device__ __forceinline__ void tdm_load_2d(const __bf16* gtile, u32 lds_off,
                                            u32 row_len_elems, u32 nrows_total,
                                            u32 row_stride_elems,
                                            u32 tile_k_elems, u32 tile_rows)
{
    unsigned long long ga = (unsigned long long)(uintptr_t)gtile;
    u32x4 g0;
    g0.x = 1u;                                   // count=1, user descriptor
    g0.y = lds_off;                              // lds_addr (wave-relative bytes)
    g0.z = (u32)ga;                              // global_addr[31:0]
    g0.w = (u32)(ga >> 32) | (2u << 30);         // global_addr[56:32] | type=2
    i32x8 g1;
    g1[0] = (int)(1u << 16);                                     // data_size=1 (2 bytes)
    g1[1] = (int)((row_len_elems & 0xffffu) << 16);              // tensor_dim0[15:0]
    g1[2] = (int)((row_len_elems >> 16) |
                  ((nrows_total & 0xffffu) << 16));              // td0 hi | tensor_dim1 lo
    g1[3] = (int)((nrows_total >> 16) | (tile_k_elems << 16));   // td1 hi | tile_dim0
    g1[4] = (int)(tile_rows & 0xffffu);                          // tile_dim1; tile_dim2=0 (2D)
    g1[5] = (int)row_stride_elems;                               // tensor_dim0_stride[31:0]
    g1[6] = 0;                                                   // stride hi | dim1_stride lo
    g1[7] = 0;                                                   // dim1_stride hi
    i32x4 g2;
    g2[0] = 1; g2[1] = 1; g2[2] = 0; g2[3] = 0;  // tensor_dim2=1, tensor_dim3=1, tile_dim3=0
    i32x4 g3 = (i32x4){0, 0, 0, 0};
    i32x8 g4 = (i32x8){0, 0, 0, 0, 0, 0, 0, 0};  // extra SGPR group (unused)
    __builtin_amdgcn_tensor_load_to_lds(g0, g1, g2, g3, g4, 0);
}
#endif

// ---------------------------------------------------------------------------
// Kernel 1: harmonics complex64 -> planar bf16 (Yr, -Yi) and transposed copies
// ---------------------------------------------------------------------------
__global__ void prep_harm(const float2* __restrict__ harm,
                          __bf16* __restrict__ Yr,   __bf16* __restrict__ Yin,
                          __bf16* __restrict__ YrT,  __bf16* __restrict__ YinT)
{
    __shared__ float sr[32][33];
    __shared__ float si[32][33];
    const int tx = threadIdx.x, ty = threadIdx.y;
    const int lm0 = blockIdx.x * 32;           // 63 blocks
    const int jk0 = blockIdx.y * 32;           // 128 blocks
    const int lm = lm0 + ty, jk = jk0 + tx;

    float2 h = harm[lm * JK + jk];
    sr[ty][tx] = h.x;
    si[ty][tx] = h.y;
    Yr [lm * JK + jk] = (__bf16)h.x;
    Yin[lm * JK + jk] = (__bf16)(-h.y);        // conj folded in

    __syncthreads();

    const int orow = jk0 + ty, ocol = lm0 + tx;
    YrT [orow * LM + ocol] = (__bf16)sr[tx][ty];
    YinT[orow * LM + ocol] = (__bf16)(-si[tx][ty]);
}

// ---------------------------------------------------------------------------
// Kernel 2: Xw[bc][jk] = bf16( a[j] * x[bc][jk] )
// ---------------------------------------------------------------------------
__global__ void prep_x(const float* __restrict__ x, const float* __restrict__ a,
                       __bf16* __restrict__ Xw)
{
    const int idx = blockIdx.x * blockDim.x + threadIdx.x;   // BC*JK threads
    const int jk  = idx & (JK - 1);
    const int j   = jk >> 6;
    Xw[idx] = (__bf16)(a[j] * x[idx]);
}

// ---------------------------------------------------------------------------
// WMMA fragment loaders (wave32, 16x16x32 bf16, contiguous k = 16*half + s in A and B).
// ---------------------------------------------------------------------------
__device__ __forceinline__ v16bf load_a_frag(const __bf16* __restrict__ base,
                                             int mrow0, int stride, int k0, int lane)
{
    const int half = lane >> 4;
    const int m    = lane & 15;
    const __bf16* p = base + (mrow0 + m) * stride;
    return *(const v16bf*)(p + k0 + 16 * half);
}

__device__ __forceinline__ v16bf load_b_frag_g(const __bf16* __restrict__ base,
                                               int nrow0, int stride, int k0, int lane)
{
    const int half = lane >> 4;
    const int n    = lane & 15;
    return *(const v16bf*)(base + (nrow0 + n) * stride + k0 + 16 * half);
}

// ---------------------------------------------------------------------------
// Kernel 3: forward SHT.  Cr[bc][lm] = sum_jk Xw*Yr ; Ci = sum_jk Xw*(-Yi).
// 8 waves / block: wave = M-tile, block = N-tile. B tiles staged to LDS by
// the Tensor Data Mover, double-buffered on TENSORcnt; every wave reuses them.
// ---------------------------------------------------------------------------
#define KC_F   4
#define CHNK_F (32 * KC_F)    // 128 K-elems per stage; 4096/128 = 32 stages

__global__ void gemm_fwd(const __bf16* __restrict__ Xw,
                         const __bf16* __restrict__ Yr,
                         const __bf16* __restrict__ Yin,
                         float* __restrict__ Cr, float* __restrict__ Ci)
{
    const int lane  = threadIdx.x & 31;
    const int wave  = threadIdx.x >> 5;
    const int mrow0 = wave * 16;          // 8 tiles cover BC=128
    const int nrow0 = blockIdx.x * 16;    // 126 blocks cover LM=2016

    v8f accR = {};
    v8f accI = {};

#if USE_TDM
    __shared__ __align__(128) __bf16 sBr[2][16 * CHNK_F];
    __shared__ __align__(128) __bf16 sBi[2][16 * CHNK_F];

    if (wave == 0) {
        tdm_load_2d(Yr  + nrow0 * JK, (u32)(uintptr_t)&sBr[0][0], JK, LM, JK, CHNK_F, 16);
        tdm_load_2d(Yin + nrow0 * JK, (u32)(uintptr_t)&sBi[0][0], JK, LM, JK, CHNK_F, 16);
    }
    int buf = 0;
    for (int kc = 0; kc < JK; kc += CHNK_F, buf ^= 1) {
        if (wave == 0) {
            if (kc + CHNK_F < JK) {
                tdm_load_2d(Yr  + nrow0 * JK + kc + CHNK_F,
                            (u32)(uintptr_t)&sBr[buf ^ 1][0], JK, LM, JK, CHNK_F, 16);
                tdm_load_2d(Yin + nrow0 * JK + kc + CHNK_F,
                            (u32)(uintptr_t)&sBi[buf ^ 1][0], JK, LM, JK, CHNK_F, 16);
                __builtin_amdgcn_s_wait_tensorcnt(2);   // current buf's pair done
            } else {
                __builtin_amdgcn_s_wait_tensorcnt(0);
            }
        }
        __syncthreads();                                // publish buf to all waves
        const int bo = (lane & 15) * CHNK_F + 16 * (lane >> 4);
        #pragma unroll
        for (int s = 0; s < KC_F; ++s) {
            v16bf af = load_a_frag(Xw, mrow0, JK, kc + s * 32, lane);
            v16bf br = *(const v16bf*)&sBr[buf][bo + s * 32];
            v16bf bi = *(const v16bf*)&sBi[buf][bo + s * 32];
            accR = __builtin_amdgcn_wmma_f32_16x16x32_bf16(false, af, false, br,
                                                           (short)0, accR, false, false);
            accI = __builtin_amdgcn_wmma_f32_16x16x32_bf16(false, af, false, bi,
                                                           (short)0, accI, false, false);
        }
        __syncthreads();                                // buf may be overwritten next stage
    }
#else
    #pragma unroll 2
    for (int k0 = 0; k0 < JK; k0 += 32) {
        v16bf af = load_a_frag(Xw, mrow0, JK, k0, lane);
        v16bf br = load_b_frag_g(Yr,  nrow0, JK, k0, lane);
        v16bf bi = load_b_frag_g(Yin, nrow0, JK, k0, lane);
        accR = __builtin_amdgcn_wmma_f32_16x16x32_bf16(false, af, false, br,
                                                       (short)0, accR, false, false);
        accI = __builtin_amdgcn_wmma_f32_16x16x32_bf16(false, af, false, bi,
                                                       (short)0, accI, false, false);
    }
#endif

    const int half = lane >> 4;
    const int col  = nrow0 + (lane & 15);
    #pragma unroll
    for (int r = 0; r < 8; ++r) {
        const int row = mrow0 + 8 * half + r;
        Cr[row * LM + col] = accR[r];
        Ci[row * LM + col] = accI[r];
    }
}

// ---------------------------------------------------------------------------
// Kernel 4: per-degree complex channel mix + mult[l] = sqrt(pi*l)
// ---------------------------------------------------------------------------
__global__ void mix(const float* __restrict__ Cr, const float* __restrict__ Ci,
                    const float2* __restrict__ w,
                    __bf16* __restrict__ Dr, __bf16* __restrict__ Di)
{
    const int idx = blockIdx.x * blockDim.x + threadIdx.x;   // BD*LM threads
    const int lm  = idx % LM;
    const int row = idx / LM;                                // bd
    const int b = row >> 4, d = row & 15;
    const int l = lm / M_M;
    const float mult = sqrtf(PI_F * (float)l);               // == 2pi/sqrt(4pi/l)

    float dr = 0.f, di = 0.f;
    #pragma unroll
    for (int c = 0; c < CIN; ++c) {
        float2 wv = w[(d * CIN + c) * L_BW + l];
        float cr = Cr[(b * CIN + c) * LM + lm];
        float ci = Ci[(b * CIN + c) * LM + lm];
        dr += wv.x * cr - wv.y * ci;
        di += wv.x * ci + wv.y * cr;
    }
    Dr[idx] = (__bf16)(mult * dr);
    Di[idx] = (__bf16)(mult * di);
}

// ---------------------------------------------------------------------------
// Kernel 5: inverse SHT.  y[bd][jk] = sum_lm Dr*Yr + Di*(-Yi) + bias[d]
// Same TDM-staged structure; both WMMAs chain into ONE f32 accumulator.
// ---------------------------------------------------------------------------
#define KC_I   3
#define CHNK_I (32 * KC_I)    // 96 K-elems per stage; 2016/96 = 21 stages

__global__ void gemm_inv(const __bf16* __restrict__ Dr, const __bf16* __restrict__ Di,
                         const __bf16* __restrict__ YrT, const __bf16* __restrict__ YinT,
                         const float* __restrict__ bias, float* __restrict__ out)
{
    const int lane  = threadIdx.x & 31;
    const int wave  = threadIdx.x >> 5;
    const int mrow0 = wave * 16;          // 8 tiles cover BD=128
    const int nrow0 = blockIdx.x * 16;    // 256 blocks cover JK=4096

    v8f acc = {};

#if USE_TDM
    __shared__ __align__(128) __bf16 sBr[2][16 * CHNK_I];
    __shared__ __align__(128) __bf16 sBi[2][16 * CHNK_I];

    if (wave == 0) {
        tdm_load_2d(YrT  + nrow0 * LM, (u32)(uintptr_t)&sBr[0][0], LM, JK, LM, CHNK_I, 16);
        tdm_load_2d(YinT + nrow0 * LM, (u32)(uintptr_t)&sBi[0][0], LM, JK, LM, CHNK_I, 16);
    }
    int buf = 0;
    for (int kc = 0; kc < LM; kc += CHNK_I, buf ^= 1) {
        if (wave == 0) {
            if (kc + CHNK_I < LM) {
                tdm_load_2d(YrT  + nrow0 * LM + kc + CHNK_I,
                            (u32)(uintptr_t)&sBr[buf ^ 1][0], LM, JK, LM, CHNK_I, 16);
                tdm_load_2d(YinT + nrow0 * LM + kc + CHNK_I,
                            (u32)(uintptr_t)&sBi[buf ^ 1][0], LM, JK, LM, CHNK_I, 16);
                __builtin_amdgcn_s_wait_tensorcnt(2);
            } else {
                __builtin_amdgcn_s_wait_tensorcnt(0);
            }
        }
        __syncthreads();
        const int bo = (lane & 15) * CHNK_I + 16 * (lane >> 4);
        #pragma unroll
        for (int s = 0; s < KC_I; ++s) {
            v16bf ar = load_a_frag(Dr, mrow0, LM, kc + s * 32, lane);
            v16bf ai = load_a_frag(Di, mrow0, LM, kc + s * 32, lane);
            v16bf br = *(const v16bf*)&sBr[buf][bo + s * 32];
            v16bf bi = *(const v16bf*)&sBi[buf][bo + s * 32];
            acc = __builtin_amdgcn_wmma_f32_16x16x32_bf16(false, ar, false, br,
                                                          (short)0, acc, false, false);
            acc = __builtin_amdgcn_wmma_f32_16x16x32_bf16(false, ai, false, bi,
                                                          (short)0, acc, false, false);
        }
        __syncthreads();
    }
#else
    #pragma unroll 2
    for (int k0 = 0; k0 < LM; k0 += 32) {
        v16bf ar = load_a_frag(Dr, mrow0, LM, k0, lane);
        v16bf ai = load_a_frag(Di, mrow0, LM, k0, lane);
        v16bf br = load_b_frag_g(YrT,  nrow0, LM, k0, lane);
        v16bf bi = load_b_frag_g(YinT, nrow0, LM, k0, lane);
        acc = __builtin_amdgcn_wmma_f32_16x16x32_bf16(false, ar, false, br,
                                                      (short)0, acc, false, false);
        acc = __builtin_amdgcn_wmma_f32_16x16x32_bf16(false, ai, false, bi,
                                                      (short)0, acc, false, false);
    }
#endif

    const int half = lane >> 4;
    const int col  = nrow0 + (lane & 15);
    #pragma unroll
    for (int r = 0; r < 8; ++r) {
        const int row = mrow0 + 8 * half + r;                // bd
        out[row * JK + col] = acc[r] + bias[row & 15];
    }
}

// ---------------------------------------------------------------------------
// Launch. Inputs: x(0), a(1), harmonics(2), weights(3), bias(4).
// Workspace (~70 MB): Yr, -Yi, YrT, -YiT (bf16), Xw (bf16),
//                     Cr, Ci (f32), Dr, Di (bf16).
// ---------------------------------------------------------------------------
extern "C" void kernel_launch(void* const* d_in, const int* in_sizes, int n_in,
                              void* d_out, int out_size, void* d_ws, size_t ws_size,
                              hipStream_t stream)
{
    const float*  x    = (const float*) d_in[0];
    const float*  a    = (const float*) d_in[1];
    const float2* harm = (const float2*)d_in[2];
    const float2* wts  = (const float2*)d_in[3];
    const float*  bias = (const float*) d_in[4];
    float* out = (float*)d_out;

    size_t off = 0;
    auto alloc = [&](size_t bytes) -> void* {
        void* p = (char*)d_ws + off;
        off += (bytes + 255) & ~(size_t)255;
        return p;
    };
    __bf16* Yr   = (__bf16*)alloc((size_t)LM * JK * 2);
    __bf16* Yin  = (__bf16*)alloc((size_t)LM * JK * 2);
    __bf16* YrT  = (__bf16*)alloc((size_t)JK * LM * 2);
    __bf16* YinT = (__bf16*)alloc((size_t)JK * LM * 2);
    __bf16* Xw   = (__bf16*)alloc((size_t)BC * JK * 2);
    float*  Cr   = (float*) alloc((size_t)BC * LM * 4);
    float*  Ci   = (float*) alloc((size_t)BC * LM * 4);
    __bf16* Dr   = (__bf16*)alloc((size_t)BD * LM * 2);
    __bf16* Di   = (__bf16*)alloc((size_t)BD * LM * 2);

    prep_harm<<<dim3(LM / 32, JK / 32), dim3(32, 32), 0, stream>>>(harm, Yr, Yin, YrT, YinT);
    prep_x<<<(BC * JK) / 256, 256, 0, stream>>>(x, a, Xw);
    gemm_fwd<<<LM / 16, 256, 0, stream>>>(Xw, Yr, Yin, Cr, Ci);
    mix<<<(BD * LM) / 256, 256, 0, stream>>>(Cr, Ci, wts, Dr, Di);
    gemm_inv<<<JK / 16, 256, 0, stream>>>(Dr, Di, YrT, YinT, bias, out);
}
